// cosine_al_seqNet_54700703482461
// MI455X (gfx1250) — matrix operand order repaired and checked by reading, not compiled, not run on hardware
//
#include <hip/hip_runtime.h>

#define Dd 4096
#define Bb 512
#define Ll 10
#define CTR 6          // L//2 + 1
#define EPSV 1e-8f

typedef __attribute__((ext_vector_type(16))) __bf16 v16bf;
typedef __attribute__((ext_vector_type(8)))  __bf16 v8bf;
typedef __attribute__((ext_vector_type(8)))  float  v8f;
typedef __attribute__((ext_vector_type(4)))  unsigned int u32x4;
typedef __attribute__((ext_vector_type(8)))  int i32x8;
typedef __attribute__((ext_vector_type(4)))  int i32x4;
typedef unsigned short u16;
typedef unsigned int   u32;

#if defined(__has_builtin)
#if __has_builtin(__builtin_amdgcn_tensor_load_to_lds) && __has_builtin(__builtin_amdgcn_s_wait_tensorcnt)
#define HAVE_TDM 1
#endif
#endif
#ifndef HAVE_TDM
#define HAVE_TDM 0
#endif

__device__ __forceinline__ u16 f2bf(float f) {
    union { float f; u32 u; } v; v.f = f;
    u32 u = v.u;
    u += 0x7fffu + ((u >> 16) & 1u);   // round-to-nearest-even
    return (u16)(u >> 16);
}
__device__ __forceinline__ float bf2f(u16 h) {
    union { u32 u; float f; } v; v.u = ((u32)h) << 16;
    return v.f;
}

// ---------------------------------------------------------------------------
// 1) Per-row stats: dot(x[b,l], x[b,CTR]) and ||x[b,l]||^2
// ---------------------------------------------------------------------------
__global__ __launch_bounds__(256) void k_stats(const float* __restrict__ x,
                                               float* __restrict__ dotb,
                                               float* __restrict__ nrm2) {
    int bl = blockIdx.x;
    int b  = bl / Ll;
    const float* row = x + (size_t)bl * Dd;
    const float* cen = x + ((size_t)b * Ll + CTR) * Dd;
    float sd = 0.f, sn = 0.f;
    for (int d = threadIdx.x; d < Dd; d += 256) {
        float xv = row[d];
        sd += xv * cen[d];
        sn += xv * xv;
    }
    __shared__ float s1[256], s2[256];
    s1[threadIdx.x] = sd; s2[threadIdx.x] = sn;
    __syncthreads();
    for (int s = 128; s > 0; s >>= 1) {
        if ((int)threadIdx.x < s) {
            s1[threadIdx.x] += s1[threadIdx.x + s];
            s2[threadIdx.x] += s2[threadIdx.x + s];
        }
        __syncthreads();
    }
    if (threadIdx.x == 0) { dotb[bl] = s1[0]; nrm2[bl] = s2[0]; }
}

// ---------------------------------------------------------------------------
// 2) xw = x * cosine_weight, converted to bf16
// ---------------------------------------------------------------------------
__global__ __launch_bounds__(256) void k_xw(const float* __restrict__ x,
                                            const float* __restrict__ dotb,
                                            const float* __restrict__ nrm2,
                                            u16* __restrict__ xw) {
    int bl = blockIdx.x;
    int b  = bl / Ll;
    float xn = fmaxf(sqrtf(nrm2[bl]), EPSV);
    float cn = fmaxf(sqrtf(nrm2[b * Ll + CTR]), EPSV);
    float w  = dotb[bl] / (xn * cn);
    const float* row = x + (size_t)bl * Dd;
    u16* orow = xw + (size_t)bl * Dd;
    for (int d = threadIdx.x; d < Dd; d += 256)
        orow[d] = f2bf(row[d] * w);
}

// ---------------------------------------------------------------------------
// 3) WT[n,k] = bf16(W[k,n])
// ---------------------------------------------------------------------------
__global__ __launch_bounds__(256) void k_transpose(const float* __restrict__ W,
                                                   u16* __restrict__ WT) {
    __shared__ float tile[32][33];
    int n0 = blockIdx.x * 32;
    int k0 = blockIdx.y * 32;
    int tx = threadIdx.x & 31, ty = threadIdx.x >> 5;   // 32 x 8
    #pragma unroll
    for (int i = 0; i < 32; i += 8)
        tile[ty + i][tx] = W[(size_t)(k0 + ty + i) * Dd + n0 + tx];
    __syncthreads();
    #pragma unroll
    for (int i = 0; i < 32; i += 8)
        WT[(size_t)(n0 + ty + i) * Dd + k0 + tx] = f2bf(tile[tx][ty + i]);
}

// ---------------------------------------------------------------------------
// TDM: one tensor_load_to_lds moves a 128x32 bf16 tile (row stride K elems)
// into LDS with 16B padding after every 64B row -> 80B LDS row stride.
// D# layout per CDNA5 ISA ch.8 (group0: count/lds/global/type, group1: dims).
// ---------------------------------------------------------------------------
#define LDSTRIDE 40   // 40 bf16 = 80 bytes per LDS row

#if HAVE_TDM
__device__ __forceinline__ void tdm_load_tile(const u16* gptr, unsigned lds_byte_off,
                                              int rows_total, int K) {
    unsigned long long ga = (unsigned long long)(size_t)gptr;
    u32x4 g0;
    g0[0] = 1u;                                              // count=1, user mode
    g0[1] = lds_byte_off;                                    // lds_addr
    g0[2] = (u32)ga;                                         // global_addr[31:0]
    g0[3] = (u32)((ga >> 32) & 0x1FFFFFFu) | (2u << 30);     // global_addr[56:32], type=2
    i32x8 g1;
    g1[0] = (int)((1u << 16) | (1u << 20) | (3u << 22) | (3u << 25));
            // data_size=2B | pad_enable | pad_interval=16DW | pad_amount=4DW
    g1[1] = (int)(((u32)K & 0xFFFFu) << 16);                 // tensor_dim0[15:0]
    g1[2] = (int)((((u32)K >> 16) & 0xFFFFu) |
                  (((u32)rows_total & 0xFFFFu) << 16));      // td0[31:16] | td1[15:0]
    g1[3] = (int)(((((u32)rows_total) >> 16) & 0xFFFFu) |
                  (32u << 16));                              // td1[31:16] | tile_dim0=32
    g1[4] = 128;                                             // tile_dim1=128, tile_dim2=0
    g1[5] = K;                                               // tensor_dim0_stride[31:0]
    g1[6] = 0;                                               // stride[47:32] | dim1_stride lo
    g1[7] = 0;
    i32x4 z4 = {0, 0, 0, 0};
    i32x8 z8 = {0, 0, 0, 0, 0, 0, 0, 0};
    // 6-arg form (clang-23 / therock-10.0 lane): trailing i32x8 is zero-filled.
    __builtin_amdgcn_tensor_load_to_lds(g0, g1, z4, z4, z8, 0);
}
#endif

// ---------------------------------------------------------------------------
// 4) WMMA GEMM: C[M,N] = A[M,K] @ BT[N,K]^T + bias  (A,BT bf16; C f32)
//    128x128 block tile, BK=32, 8 waves (4Mx2N), 8 v_wmma per wave per K-step.
//    Double-buffered LDS fed by the Tensor Data Mover (TENSORcnt-tracked).
// ---------------------------------------------------------------------------
__global__ __launch_bounds__(256) void k_gemm(const u16* __restrict__ A,
                                              const u16* __restrict__ BT,
                                              const float* __restrict__ bias,
                                              float* __restrict__ C,
                                              int M, int N, int K) {
    __shared__ __bf16 As[2][128][LDSTRIDE];
    __shared__ __bf16 Bs[2][128][LDSTRIDE];

    const int nb0  = blockIdx.x * 128;
    const int mb0  = blockIdx.y * 128;
    const int tid  = threadIdx.x;
    const int wave = tid >> 5;
    const int lane = tid & 31;
    const int wm   = (wave >> 1) * 32;   // 0,32,64,96
    const int wn   = (wave & 1) * 64;    // 0,64

    v8f acc[2][4];
    #pragma unroll
    for (int i = 0; i < 2; i++)
        #pragma unroll
        for (int j = 0; j < 4; j++)
            #pragma unroll
            for (int r = 0; r < 8; r++) acc[i][j][r] = 0.f;

    const u16* Abase = A  + (size_t)mb0 * K;
    const u16* Bbase = BT + (size_t)nb0 * K;

    const int frow  = lane & 15;
    const int akoff = (lane < 16) ? 0 : 8;
    const int bkoff = (lane < 16) ? 0 : 8;
    const int nsteps = K / 32;

#if HAVE_TDM
    const unsigned offA[2] = { (unsigned)(size_t)&As[0][0][0], (unsigned)(size_t)&As[1][0][0] };
    const unsigned offB[2] = { (unsigned)(size_t)&Bs[0][0][0], (unsigned)(size_t)&Bs[1][0][0] };
    if (wave == 0) {
        tdm_load_tile(Abase, offA[0], M, K);
        tdm_load_tile(Bbase, offB[0], N, K);
    }
#else
    const int lr = tid >> 1;
    const int lc = (tid & 1) * 16;
    const u16* Ap = A  + (size_t)(mb0 + lr) * K + lc;
    const u16* Bp = BT + (size_t)(nb0 + lr) * K + lc;
#endif

    for (int s = 0; s < nsteps; ++s) {
        const int cur = s & 1;
#if HAVE_TDM
        if (wave == 0) {
            if (s + 1 < nsteps) {
                const int nxt = cur ^ 1;
                tdm_load_tile(Abase + (size_t)(s + 1) * 32, offA[nxt], M, K);
                tdm_load_tile(Bbase + (size_t)(s + 1) * 32, offB[nxt], N, K);
                __builtin_amdgcn_s_wait_tensorcnt(2);   // oldest pair (cur) done
            } else {
                __builtin_amdgcn_s_wait_tensorcnt(0);
            }
        }
        __syncthreads();
#else
        {
            const int k0 = s * 32;
            const uint4* ag = (const uint4*)(Ap + k0);
            *(uint4*)&As[cur][lr][lc]     = ag[0];
            *(uint4*)&As[cur][lr][lc + 8] = ag[1];
            const uint4* bg = (const uint4*)(Bp + k0);
            *(uint4*)&Bs[cur][lr][lc]     = bg[0];
            *(uint4*)&Bs[cur][lr][lc + 8] = bg[1];
        }
        __syncthreads();
#endif
        v16bf afrag[2];
        #pragma unroll
        for (int i = 0; i < 2; i++) {
            v8bf lo = *(const v8bf*)&As[cur][wm + i * 16 + frow][akoff];
            v8bf hi = *(const v8bf*)&As[cur][wm + i * 16 + frow][akoff + 16];
            #pragma unroll
            for (int e = 0; e < 8; e++) { afrag[i][e] = lo[e]; afrag[i][e + 8] = hi[e]; }
        }
        v16bf bfrag[4];
        #pragma unroll
        for (int j = 0; j < 4; j++) {
            v8bf lo = *(const v8bf*)&Bs[cur][wn + j * 16 + frow][bkoff];
            v8bf hi = *(const v8bf*)&Bs[cur][wn + j * 16 + frow][bkoff + 16];
            #pragma unroll
            for (int e = 0; e < 8; e++) { bfrag[j][e] = lo[e]; bfrag[j][e + 8] = hi[e]; }
        }

        #pragma unroll
        for (int i = 0; i < 2; i++)
            #pragma unroll
            for (int j = 0; j < 4; j++)
                acc[i][j] = __builtin_amdgcn_wmma_f32_16x16x32_bf16(
                    false, afrag[i], false, bfrag[j],
                    (short)0, acc[i][j], false, false);
        __syncthreads();   // all reads of buffer `cur` done before it is refilled
    }

    const int crow  = lane & 15;
    const int chalf = (lane < 16) ? 0 : 8;
    #pragma unroll
    for (int i = 0; i < 2; i++) {
        #pragma unroll
        for (int j = 0; j < 4; j++) {
            int gm = mb0 + wm + i * 16 + chalf;
            int gn = nb0 + wn + j * 16 + crow;
            float bvv = bias[gn];
            #pragma unroll
            for (int r = 0; r < 8; r++)
                C[(size_t)(gm + r) * N + gn] = acc[i][j][r] + bvv;
        }
    }
}

// ---------------------------------------------------------------------------
// 5) Attention per batch + y[b,:] = sum_m c[m]*xw[b,m,:]
// ---------------------------------------------------------------------------
__global__ __launch_bounds__(256) void k_attn(const float* __restrict__ q,
                                              const float* __restrict__ k,
                                              const u16* __restrict__ xw,
                                              u16* __restrict__ y) {
    int b = blockIdx.x;
    __shared__ float sc[Ll][Ll];
    __shared__ float cc[Ll];
    int tid = threadIdx.x, wave = tid >> 5, lane = tid & 31;
    const float* qb = q + (size_t)b * Ll * Dd;
    const float* kb = k + (size_t)b * Ll * Dd;

    for (int p = wave; p < Ll * Ll; p += 8) {
        int l = p / Ll, m = p % Ll;
        const float* qr = qb + (size_t)l * Dd;
        const float* kr = kb + (size_t)m * Dd;
        float s = 0.f;
        for (int d = lane; d < Dd; d += 32) s += qr[d] * kr[d];
        #pragma unroll
        for (int off = 16; off > 0; off >>= 1) s += __shfl_xor(s, off, 32);
        if (lane == 0) sc[l][m] = s;
    }
    __syncthreads();
    if (tid < Ll) {
        int l = tid;
        float mx = sc[l][0];
        for (int m = 1; m < Ll; m++) mx = fmaxf(mx, sc[l][m]);
        float e[Ll]; float sum = 0.f;
        for (int m = 0; m < Ll; m++) { e[m] = expf(sc[l][m] - mx); sum += e[m]; }
        for (int m = 0; m < Ll; m++) sc[l][m] = e[m] / sum;
    }
    __syncthreads();
    if (tid < Ll) {
        int m = tid; float s = 0.f;
        for (int l = 0; l < Ll; l++) s += sc[l][m];
        cc[m] = s * (1.0f / Ll);
    }
    __syncthreads();
    const u16* xb = xw + (size_t)b * Ll * Dd;
    u16* yb = y + (size_t)b * Dd;
    for (int d = tid; d < Dd; d += 256) {
        float s = 0.f;
        #pragma unroll
        for (int m = 0; m < Ll; m++) s += cc[m] * bf2f(xb[(size_t)m * Dd + d]);
        yb[d] = f2bf(s);
    }
}

// ---------------------------------------------------------------------------
extern "C" void kernel_launch(void* const* d_in, const int* in_sizes, int n_in,
                              void* d_out, int out_size, void* d_ws, size_t ws_size,
                              hipStream_t stream) {
    const float* x  = (const float*)d_in[0];
    const float* Wq = (const float*)d_in[1];
    const float* bq = (const float*)d_in[2];
    const float* Wk = (const float*)d_in[3];
    const float* bk = (const float*)d_in[4];
    const float* Wv = (const float*)d_in[5];
    const float* bv = (const float*)d_in[6];
    float* out = (float*)d_out;

    size_t off = 0;
    char* base = (char*)d_ws;
    auto carve = [&](size_t bytes) -> void* {
        void* p = (void*)(base + off);
        off += (bytes + 255) & ~(size_t)255;
        return p;
    };
    const size_t BL = (size_t)Bb * Ll;
    float* dotb = (float*)carve(BL * 4);
    float* nrm2 = (float*)carve(BL * 4);
    u16*   xw   = (u16*)  carve(BL * Dd * 2);
    u16*   WqT  = (u16*)  carve((size_t)Dd * Dd * 2);
    u16*   WkT  = (u16*)  carve((size_t)Dd * Dd * 2);
    u16*   WvT  = (u16*)  carve((size_t)Dd * Dd * 2);
    float* qbuf = (float*)carve(BL * Dd * 4);
    float* kbuf = (float*)carve(BL * Dd * 4);
    u16*   ybuf = (u16*)  carve((size_t)Bb * Dd * 2);

    k_stats<<<(int)BL, 256, 0, stream>>>(x, dotb, nrm2);
    k_xw   <<<(int)BL, 256, 0, stream>>>(x, dotb, nrm2, xw);

    dim3 tg(Dd / 32, Dd / 32);
    k_transpose<<<tg, 256, 0, stream>>>(Wq, WqT);
    k_transpose<<<tg, 256, 0, stream>>>(Wk, WkT);
    k_transpose<<<tg, 256, 0, stream>>>(Wv, WvT);

    dim3 gqk(Dd / 128, (int)(BL / 128));           // 32 x 40
    k_gemm<<<gqk, 256, 0, stream>>>(xw, WqT, bq, qbuf, (int)BL, Dd, Dd);
    k_gemm<<<gqk, 256, 0, stream>>>(xw, WkT, bk, kbuf, (int)BL, Dd, Dd);

    k_attn<<<Bb, 256, 0, stream>>>(qbuf, kbuf, xw, ybuf);

    dim3 gv(Dd / 128, Bb / 128);                   // 32 x 4
    k_gemm<<<gv, 256, 0, stream>>>(ybuf, WvT, bv, out, Bb, Dd, Dd);
}
